// VisionEncoderLayer_2619930051235
// MI455X (gfx1250) — compile-verified, matrix operand order checked by reading, not run on hardware
//
#include <hip/hip_runtime.h>
#include <hip/hip_bf16.h>

// ---------------------------------------------------------------------------
// ViT encoder layer on MI455X (gfx1250, wave32, WMMA).
// All GEMMs: f16 inputs, f32 accumulate via v_wmma_f32_16x16x32_f16.
// GEMM tile staging: double-buffered CDNA5 async global->LDS DMA (ASYNCcnt),
// draining with s_wait_asynccnt 3 so the next tile stays in flight.
// ---------------------------------------------------------------------------

#define H   768
#define NH  12
#define HD  64
#define IS  3072
#define BATCH 16
#define SEQ 1024
#define NTOK (BATCH * SEQ)

#define USE_ASYNC_LDS 1   // flip to 0 for the synchronous VGPR-staging fallback

typedef __attribute__((ext_vector_type(16))) _Float16 v16h;
typedef __attribute__((ext_vector_type(8)))  _Float16 v8h;
typedef __attribute__((ext_vector_type(8)))  float    v8f;

// --------------------------- CDNA5 async copy helpers ----------------------
// GLOBAL_LOAD_ASYNC_TO_LDS_B128 (GV mode): per-lane 16B global->LDS DMA,
// tracked by ASYNCcnt. VDST VGPR carries the LDS byte address (addr[31:0] of
// the flat shared pointer), VADDR the 64-bit global address.

__device__ __forceinline__ void async_copy_b128(unsigned lds_addr, const void* gptr) {
  asm volatile("global_load_async_to_lds_b128 %0, %1, off"
               :: "v"(lds_addr), "v"(gptr) : "memory");
}
// Async loads complete in order: cnt<=3 means everything but the 3 most
// recently issued copies has landed in LDS.
__device__ __forceinline__ void wait_async3() {
  asm volatile("s_wait_asynccnt 0x3" ::: "memory");
}
__device__ __forceinline__ void wait_async0() {
  asm volatile("s_wait_asynccnt 0x0" ::: "memory");
}
__device__ __forceinline__ unsigned lds_addr_of(const void* p) {
  return (unsigned)(unsigned long long)p;  // flat LDS address: addr[31:0] = LDS offset
}

// --------------------------- WMMA helpers ----------------------------------

__device__ __forceinline__ v8f wmma32(v16h a, v16h b, v8f c) {
  return __builtin_amdgcn_wmma_f32_16x16x32_f16(
      /*neg_a=*/false, a, /*neg_b=*/false, b,
      /*c_mod=*/(short)0, c, /*reuse_a=*/false, /*reuse_b=*/false);
}

// Load a 16x32 f16 fragment (A or B operand; layouts are symmetric) from a
// row-major matrix with leading dimension `ld` (in elements).
// ISA 7.12.2: lanes 0-15 hold row (lane), K = 0..7 (vgpr0-3) and 16..23
// (vgpr4-7); lanes 16-31 hold the same rows with K = 8..15 and 24..31.
__device__ __forceinline__ v16h load_frag(const _Float16* __restrict__ base, int ld) {
  const int lane = threadIdx.x & 31;
  const int r    = lane & 15;
  const int hi   = lane >> 4;
  const _Float16* p = base + (size_t)r * ld + hi * 8;
  v8h lo = *(const v8h*)(p);       // K = 8*hi .. 8*hi+7
  v8h hk = *(const v8h*)(p + 16);  // K = 16+8*hi .. 16+8*hi+7
  v16h out;
#pragma unroll
  for (int j = 0; j < 8; ++j) { out[j] = lo[j]; out[j + 8] = hk[j]; }
  return out;
}

__device__ __forceinline__ float gelu_tanh(float x) {
  const float k0 = 0.7978845608028654f;  // sqrt(2/pi)
  const float k1 = 0.044715f;
  float x3 = x * x * x;
  return 0.5f * x * (1.f + tanhf(k0 * (x + k1 * x3)));
}

// --------------------------- small kernels ---------------------------------

__global__ __launch_bounds__(256) void f32_to_f16_k(
    const float* __restrict__ in, _Float16* __restrict__ out, int n) {
  int i = blockIdx.x * 256 + threadIdx.x;
  if (i < n) out[i] = (_Float16)in[i];
}

// One block per token; 768 = 256 * 3 elements per thread.
__global__ __launch_bounds__(256) void layernorm_f16_k(
    const float* __restrict__ x, const float* __restrict__ g,
    const float* __restrict__ b, _Float16* __restrict__ out) {
  const int t = blockIdx.x;
  const float* xr = x + (size_t)t * H;
  const int i0 = threadIdx.x, i1 = i0 + 256, i2 = i0 + 512;
  float v0 = xr[i0], v1 = xr[i1], v2 = xr[i2];
  float s  = v0 + v1 + v2;
  float s2 = v0 * v0 + v1 * v1 + v2 * v2;
#pragma unroll
  for (int m = 16; m >= 1; m >>= 1) {
    s  += __shfl_xor(s,  m);
    s2 += __shfl_xor(s2, m);
  }
  __shared__ float ps[8], ps2[8];
  const int w = threadIdx.x >> 5, lane = threadIdx.x & 31;
  if (lane == 0) { ps[w] = s; ps2[w] = s2; }
  __syncthreads();
  if (w == 0) {
    s  = (lane < 8) ? ps[lane]  : 0.f;
    s2 = (lane < 8) ? ps2[lane] : 0.f;
#pragma unroll
    for (int m = 4; m >= 1; m >>= 1) {
      s  += __shfl_xor(s,  m);
      s2 += __shfl_xor(s2, m);
    }
    if (lane == 0) { ps[0] = s; ps2[0] = s2; }
  }
  __syncthreads();
  const float mu  = ps[0] * (1.f / H);
  const float var = ps2[0] * (1.f / H) - mu * mu;
  const float inv = rsqrtf(var + 1e-6f);
  _Float16* orow = out + (size_t)t * H;
  orow[i0] = (_Float16)((v0 - mu) * inv * g[i0] + b[i0]);
  orow[i1] = (_Float16)((v1 - mu) * inv * g[i1] + b[i1]);
  orow[i2] = (_Float16)((v2 - mu) * inv * g[i2] + b[i2]);
}

// --------------------------- tiled WMMA GEMM -------------------------------
// D[M x N] = A[M x K] * W^T (+ bias) with W row-major [N x K] (torch Linear).
// Block tile 128x64, 8 waves arranged 4x2, each wave 32x32 (2x2 frags).
// Double-buffered async pipeline; K must be a multiple of 64 (768 / 3072).
// EPI: 0 = store f16, 1 = store f16 after GELU, 2 = store f32 + residual.

#define LDT 48  // LDS row stride in f16 (96B: 16B aligned, staggers banks)

template <int EPI>
__global__ __launch_bounds__(256) void gemm_wmma_k(
    const _Float16* __restrict__ A, const _Float16* __restrict__ W,
    const float* __restrict__ bias, const float* __restrict__ resid,
    void* __restrict__ out, int M, int N, int K) {
  __shared__ _Float16 sA[2][128 * LDT];
  __shared__ _Float16 sB[2][64 * LDT];

  const int m0 = blockIdx.y * 128;
  const int n0 = blockIdx.x * 64;
  const int tid = threadIdx.x;
  const int w = tid >> 5, lane = tid & 31;
  const int wm = (w >> 1) * 32;  // wave row offset in block tile
  const int wn = (w & 1) * 32;   // wave col offset

  v8f c[2][2] = {};

  const int row = tid >> 2;      // 0..63
  const int q   = tid & 3;       // 0..3 (8 f16 each)
  const unsigned la0[2] = { lds_addr_of(&sA[0][row * LDT + q * 8]),
                            lds_addr_of(&sA[1][row * LDT + q * 8]) };
  const unsigned la1[2] = { lds_addr_of(&sA[0][(row + 64) * LDT + q * 8]),
                            lds_addr_of(&sA[1][(row + 64) * LDT + q * 8]) };
  const unsigned lb0[2] = { lds_addr_of(&sB[0][row * LDT + q * 8]),
                            lds_addr_of(&sB[1][row * LDT + q * 8]) };

  const _Float16* gaRow = A + (size_t)(m0 + row) * K + q * 8;
  const _Float16* gbRow = W + (size_t)(n0 + row) * K + q * 8;

  auto issue = [&](int buf, int k) {
#if USE_ASYNC_LDS
    async_copy_b128(la0[buf], gaRow + k);
    async_copy_b128(la1[buf], gaRow + (size_t)64 * K + k);
    async_copy_b128(lb0[buf], gbRow + k);
#else
    *(v8h*)&sA[buf][row * LDT + q * 8]        = *(const v8h*)(gaRow + k);
    *(v8h*)&sA[buf][(row + 64) * LDT + q * 8] = *(const v8h*)(gaRow + (size_t)64 * K + k);
    *(v8h*)&sB[buf][row * LDT + q * 8]        = *(const v8h*)(gbRow + k);
#endif
  };
  auto compute = [&](int buf) {
    v16h af0 = load_frag(&sA[buf][(wm + 0)  * LDT], LDT);
    v16h af1 = load_frag(&sA[buf][(wm + 16) * LDT], LDT);
    v16h bf0 = load_frag(&sB[buf][(wn + 0)  * LDT], LDT);
    v16h bf1 = load_frag(&sB[buf][(wn + 16) * LDT], LDT);
    c[0][0] = wmma32(af0, bf0, c[0][0]);
    c[0][1] = wmma32(af0, bf1, c[0][1]);
    c[1][0] = wmma32(af1, bf0, c[1][0]);
    c[1][1] = wmma32(af1, bf1, c[1][1]);
  };

  issue(0, 0);                                   // prologue: tile 0 in flight
  for (int k0 = 0; k0 < K; k0 += 64) {
    // ---- stage A: compute tile (k0) from buf0, tile (k0+32) lands in buf1
    issue(1, k0 + 32);                           // always valid: K % 64 == 0
#if USE_ASYNC_LDS
    wait_async3();                               // drain tile k0, keep k0+32 in flight
    __builtin_prefetch(gaRow + k0 + 64, 0, 3);   // two tiles ahead
    __builtin_prefetch(gbRow + k0 + 64, 0, 3);
#endif
    __syncthreads();
    compute(0);
    __syncthreads();                             // buf0 free for re-issue

    // ---- stage B: compute tile (k0+32) from buf1, tile (k0+64) lands in buf0
    const bool more = (k0 + 64) < K;
    if (more) issue(0, k0 + 64);
#if USE_ASYNC_LDS
    if (more) wait_async3(); else wait_async0();
#endif
    __syncthreads();
    compute(1);
    __syncthreads();                             // buf1 free for re-issue
  }

  // Epilogue. C/D layout: lane = column (lane&15), vgpr v = row v + 8*(lane>>4).
  const int col = lane & 15, hi = lane >> 4;
#pragma unroll
  for (int i = 0; i < 2; ++i) {
#pragma unroll
    for (int j = 0; j < 2; ++j) {
      const int n = n0 + wn + j * 16 + col;
      const float bv = bias[n];
#pragma unroll
      for (int v = 0; v < 8; ++v) {
        const int m = m0 + wm + i * 16 + v + 8 * hi;
        const size_t idx = (size_t)m * N + n;
        float val = c[i][j][v] + bv;
        if (EPI == 0) {
          ((_Float16*)out)[idx] = (_Float16)val;
        } else if (EPI == 1) {
          ((_Float16*)out)[idx] = (_Float16)gelu_tanh(val);
        } else {
          ((float*)out)[idx] = resid[idx] + val;
        }
      }
    }
  }
}

// --------------------------- flash attention -------------------------------
// One block per (b, h, 128-query tile). 8 waves, each owns 16 query rows.
// Key tiles staged with async global->LDS DMA; V is staged transposed
// (manual, so the P*V B-fragment reads row-major). Online softmax with
// half-wave shfl_xor row reductions (rows 0-7 in lanes 0-15, 8-15 in 16-31).

__global__ __launch_bounds__(256) void attention_wmma_k(
    const _Float16* __restrict__ Q, const _Float16* __restrict__ K,
    const _Float16* __restrict__ V, _Float16* __restrict__ O) {
  __shared__ _Float16 sK[64 * 64];       // [key][d]
  __shared__ _Float16 sV[64 * 64];       // [d][key]  (V^T)
  __shared__ _Float16 sP[8][16 * 64];    // per-wave probability tile

  const int bid = blockIdx.x;
  const int qt  = bid & 7;               // SEQ/128 = 8 tiles
  const int bh  = bid >> 3;
  const int b   = bh / NH, h = bh % NH;
  const int tid = threadIdx.x, w = tid >> 5, lane = tid & 31;
  const int col = lane & 15, hi = lane >> 4;
  const int q0  = qt * 128 + w * 16;
  const size_t tokBase = (size_t)b * SEQ;
  const float scale = 0.125f;            // HD^-0.5

  const _Float16* qbase = Q + (tokBase + q0) * H + h * HD;
  const v16h qf0 = load_frag(qbase, H);        // d = 0..31
  const v16h qf1 = load_frag(qbase + 32, H);   // d = 32..63

  v8f of[4] = {};                        // output 16 x 64
  float rm[8], rs[8];
#pragma unroll
  for (int v = 0; v < 8; ++v) { rm[v] = -1e30f; rs[v] = 0.f; }

  const int kk = tid >> 2;               // 0..63 (staging row)
  const int q4 = tid & 3;                // 0..3  (16 f16 chunk)
  const unsigned ldsK0 = lds_addr_of(&sK[kk * 64 + q4 * 16]);
  const unsigned ldsK1 = lds_addr_of(&sK[kk * 64 + q4 * 16 + 8]);

  for (int kt = 0; kt < SEQ; kt += 64) {
    __syncthreads();                     // previous tile fully consumed
    {
      const _Float16* kg = K + (tokBase + kt + kk) * H + h * HD + q4 * 16;
#if USE_ASYNC_LDS
      async_copy_b128(ldsK0, kg);
      async_copy_b128(ldsK1, kg + 8);
#else
      *(v8h*)&sK[kk * 64 + q4 * 16]     = *(const v8h*)kg;
      *(v8h*)&sK[kk * 64 + q4 * 16 + 8] = *(const v8h*)(kg + 8);
#endif
      const _Float16* vg = V + (tokBase + kt + kk) * H + h * HD + q4 * 16;
#pragma unroll
      for (int j = 0; j < 16; ++j) sV[(q4 * 16 + j) * 64 + kk] = vg[j];
#if USE_ASYNC_LDS
      wait_async0();
#endif
    }
    __syncthreads();

    // scores S = Q * K^T : 16 x 64 in four 16x16 frags
    v8f s[4];
#pragma unroll
    for (int f = 0; f < 4; ++f) {
      v16h kb0 = load_frag(&sK[(f * 16) * 64], 64);
      v16h kb1 = load_frag(&sK[(f * 16) * 64 + 32], 64);
      v8f acc = {};
      acc = wmma32(qf0, kb0, acc);
      acc = wmma32(qf1, kb1, acc);
      s[f] = acc;
    }

    // online softmax (row m = v + 8*hi lives in one 16-lane half)
#pragma unroll
    for (int v = 0; v < 8; ++v) {
      float tmax = s[0][v] * scale;
#pragma unroll
      for (int f = 0; f < 4; ++f) {
        s[f][v] *= scale;
        tmax = fmaxf(tmax, s[f][v]);
      }
#pragma unroll
      for (int m = 8; m >= 1; m >>= 1) tmax = fmaxf(tmax, __shfl_xor(tmax, m));
      const float mnew  = fmaxf(rm[v], tmax);
      const float alpha = __expf(rm[v] - mnew);
      float tsum = 0.f;
#pragma unroll
      for (int f = 0; f < 4; ++f) {
        const float p = __expf(s[f][v] - mnew);
        s[f][v] = p;
        tsum += p;
      }
#pragma unroll
      for (int m = 8; m >= 1; m >>= 1) tsum += __shfl_xor(tsum, m);
      rs[v] = rs[v] * alpha + tsum;
      rm[v] = mnew;
#pragma unroll
      for (int j = 0; j < 4; ++j) of[j][v] *= alpha;
      // stash P into this wave's LDS tile (C/D layout -> row major)
#pragma unroll
      for (int f = 0; f < 4; ++f)
        sP[w][(v + 8 * hi) * 64 + f * 16 + col] = (_Float16)s[f][v];
    }

    // O += P * V   (same-wave LDS store->load is in program order)
#pragma unroll
    for (int ks = 0; ks < 2; ++ks) {
      v16h pa = load_frag(&sP[w][ks * 32], 64);
#pragma unroll
      for (int j = 0; j < 4; ++j) {
        v16h vb = load_frag(&sV[(j * 16) * 64 + ks * 32], 64);
        of[j] = wmma32(pa, vb, of[j]);
      }
    }
  }

  // normalize and store
#pragma unroll
  for (int j = 0; j < 4; ++j) {
#pragma unroll
    for (int v = 0; v < 8; ++v) {
      const float val = of[j][v] / rs[v];
      O[(tokBase + q0 + v + 8 * hi) * H + h * HD + j * 16 + col] = (_Float16)val;
    }
  }
}

// --------------------------- launcher --------------------------------------

extern "C" void kernel_launch(void* const* d_in, const int* in_sizes, int n_in,
                              void* d_out, int out_size, void* d_ws, size_t ws_size,
                              hipStream_t stream) {
  const float* hs    = (const float*)d_in[0];
  const float* wq    = (const float*)d_in[1];
  const float* bq    = (const float*)d_in[2];
  const float* wk    = (const float*)d_in[3];
  const float* bk    = (const float*)d_in[4];
  const float* wv    = (const float*)d_in[5];
  const float* bv    = (const float*)d_in[6];
  const float* wo    = (const float*)d_in[7];
  const float* bo    = (const float*)d_in[8];
  const float* ln1g  = (const float*)d_in[9];
  const float* ln1b  = (const float*)d_in[10];
  const float* ln2g  = (const float*)d_in[11];
  const float* ln2b  = (const float*)d_in[12];
  const float* w1    = (const float*)d_in[13];
  const float* b1    = (const float*)d_in[14];
  const float* w2    = (const float*)d_in[15];
  const float* b2    = (const float*)d_in[16];

  // workspace arena (aliased; ~266 MB peak)
  char* ws = (char*)d_ws;
  size_t off = 0;
  auto carve = [&](size_t bytes) {
    char* p = ws + off;
    off = (off + bytes + 255) & ~(size_t)255;
    return p;
  };
  const size_t SZ_HH  = (size_t)H * H * sizeof(_Float16);
  const size_t SZ_IH  = (size_t)IS * H * sizeof(_Float16);
  const size_t SZ_TH2 = (size_t)NTOK * H * sizeof(_Float16);
  const size_t SZ_TH4 = (size_t)NTOK * H * sizeof(float);
  const size_t SZ_TI2 = (size_t)NTOK * IS * sizeof(_Float16);

  _Float16* wq16 = (_Float16*)carve(SZ_HH);
  _Float16* wk16 = (_Float16*)carve(SZ_HH);
  _Float16* wv16 = (_Float16*)carve(SZ_HH);
  _Float16* wo16 = (_Float16*)carve(SZ_HH);
  _Float16* w116 = (_Float16*)carve(SZ_IH);
  _Float16* w216 = (_Float16*)carve(SZ_IH);
  _Float16* bufA = (_Float16*)carve(SZ_TH2);  // ln1 out, later attn out
  _Float16* bufB = (_Float16*)carve(SZ_TH2);  // q, later ln2 out
  _Float16* bufC = (_Float16*)carve(SZ_TH2);  // k
  _Float16* bufD = (_Float16*)carve(SZ_TH2);  // v
  float*    x2   = (float*)   carve(SZ_TH4);  // post-attention residual state
  _Float16* h1   = (_Float16*)carve(SZ_TI2);  // gelu(fc1)
  (void)ws_size; (void)n_in; (void)in_sizes; (void)out_size;

  // 1) weights -> f16
  const int nHH = H * H, nIH = IS * H;
  f32_to_f16_k<<<(nHH + 255) / 256, 256, 0, stream>>>(wq, wq16, nHH);
  f32_to_f16_k<<<(nHH + 255) / 256, 256, 0, stream>>>(wk, wk16, nHH);
  f32_to_f16_k<<<(nHH + 255) / 256, 256, 0, stream>>>(wv, wv16, nHH);
  f32_to_f16_k<<<(nHH + 255) / 256, 256, 0, stream>>>(wo, wo16, nHH);
  f32_to_f16_k<<<(nIH + 255) / 256, 256, 0, stream>>>(w1, w116, nIH);
  f32_to_f16_k<<<(nIH + 255) / 256, 256, 0, stream>>>(w2, w216, nIH);

  const dim3 blk(256);
  const dim3 gH(H / 64, NTOK / 128);    // N=768 GEMMs
  const dim3 gI(IS / 64, NTOK / 128);   // N=3072 GEMM

  // 2) LN1
  layernorm_f16_k<<<NTOK, blk, 0, stream>>>(hs, ln1g, ln1b, bufA);

  // 3) QKV projections
  gemm_wmma_k<0><<<gH, blk, 0, stream>>>(bufA, wq16, bq, nullptr, bufB, NTOK, H, H);
  gemm_wmma_k<0><<<gH, blk, 0, stream>>>(bufA, wk16, bk, nullptr, bufC, NTOK, H, H);
  gemm_wmma_k<0><<<gH, blk, 0, stream>>>(bufA, wv16, bv, nullptr, bufD, NTOK, H, H);

  // 4) attention (bufA reused as output; ln1 activations are dead now)
  attention_wmma_k<<<BATCH * NH * (SEQ / 128), blk, 0, stream>>>(bufB, bufC, bufD, bufA);

  // 5) output projection + residual -> x2 (f32)
  gemm_wmma_k<2><<<gH, blk, 0, stream>>>(bufA, wo16, bo, hs, x2, NTOK, H, H);

  // 6) LN2 (bufB reused; q is dead)
  layernorm_f16_k<<<NTOK, blk, 0, stream>>>(x2, ln2g, ln2b, bufB);

  // 7) FC1 + GELU
  gemm_wmma_k<1><<<gI, blk, 0, stream>>>(bufB, w116, b1, nullptr, h1, NTOK, IS, H);

  // 8) FC2 + residual -> d_out (f32)
  gemm_wmma_k<2><<<gH, blk, 0, stream>>>(h1, w216, b2, x2, (float*)d_out, NTOK, H, IS);
}
